// LookupFFN_59940563583085
// MI455X (gfx1250) — matrix operand order repaired and checked by reading, not compiled, not run on hardware
//
#include <hip/hip_runtime.h>
#include <hip/hip_bf16.h>

typedef __attribute__((ext_vector_type(16))) _Float16 v16h;
typedef __attribute__((ext_vector_type(8)))  float    v8f;

#define HIDDEN      1024
#define NUM_TABLE   32
#define TABLE_SIZE  1024
#define CODE_LEN    10
#define OUT_SIZE    1024
#define BLOCK       32
#define NUM_BLOCK   32
#define ROWS_TILE   16
#define DECAY       0.7f
#define LN_EPS      1e-12f

// ---------------------------------------------------------------------------
// Kernel 0: pack fp32 weights (1,4,32,32,32) into f16 in the exact WMMA
// B-operand per-lane layout, so stage 1 loads each lane's operand as one
// contiguous 32-byte read. Group g = (s*32+b)*2 + tn; lane l holds
// B(K = kb2+i (i<8) or 16+kb2+i-8 (i>=8), N = (l&15)+tn*16) with kb2 = (l<16) ? 0 : 8  (ISA 7.12.2).
// Packed size: 256 groups * 32 lanes * 16 halves = 256 KB (L2-resident).
// ---------------------------------------------------------------------------
__global__ __launch_bounds__(32) void lookupffn_pack_w(
    const float* __restrict__ w, _Float16* __restrict__ wpack)
{
    const int g    = blockIdx.x;       // 0..255 = (s*32+b)*2+tn
    const int tn   = g & 1;
    const int sb   = g >> 1;           // s*32+b
    const int lane = threadIdx.x;      // wave32
    const int n    = (lane & 15) + tn * 16;
    const int kb2  = (lane < 16) ? 0 : 8;
    const float* wp = w + (size_t)sb * BLOCK * BLOCK;
    v16h v;
    #pragma unroll
    for (int i = 0; i < 16; i++)
        v[i] = (_Float16)wp[((i < 8) ? (kb2 + i) : (16 + kb2 + i - 8)) * BLOCK + n];
    *(v16h*)(wpack + ((size_t)g * 32 + lane) * 16) = v;
}

// ---------------------------------------------------------------------------
// Kernel 1: LayerNorm + BH4 (4 x [block-diag 32x32 matmul via WMMA + FWHT])
//           + code/score extraction. One 16-row tile per workgroup, fully in
//           LDS (xn + ping/pong buffers = 192 KB of the WGP's 320 KB).
// ---------------------------------------------------------------------------
__global__ __launch_bounds__(256) void lookupffn_stage1(
    const float*    __restrict__ x,      // (N, 1024)
    const float*    __restrict__ gamma,  // (1024,)
    const float*    __restrict__ beta,   // (1024,)
    const _Float16* __restrict__ wpack,  // packed f16 B operands
    const float*    __restrict__ bias,   // (320,)
    int*   __restrict__ d_code,          // (N, 32)  code + t*TABLE_SIZE folded
    float* __restrict__ d_score)         // (N, 32)
{
    extern __shared__ float smem[];
    float* xnbuf = smem;               // 16*1024 fp32  (LayerNorm output, kept)
    float* bufA  = smem + 16 * 1024;   // ping
    float* bufB  = smem + 32 * 1024;   // pong

    const int  tid  = threadIdx.x;
    const int  wave = tid >> 5;        // 0..7  (wave32)
    const int  lane = tid & 31;
    const long row0 = (long)blockIdx.x * ROWS_TILE;

    // ---- LayerNorm: wave w owns rows 2w, 2w+1; wave-level shuffle reduce ----
    #pragma unroll
    for (int rr = 0; rr < 2; rr++) {
        const int r = wave * 2 + rr;
        const float* xr = x + (row0 + r) * HIDDEN;
        float s = 0.f, s2 = 0.f;
        for (int k = lane; k < HIDDEN; k += 32) {
            float v = xr[k];
            xnbuf[r * HIDDEN + k] = v;
            s += v; s2 += v * v;
        }
        #pragma unroll
        for (int off = 16; off; off >>= 1) {
            s  += __shfl_xor(s,  off, 32);
            s2 += __shfl_xor(s2, off, 32);
        }
        const float mu  = s * (1.0f / HIDDEN);
        const float var = s2 * (1.0f / HIDDEN) - mu * mu;
        const float inv = rsqrtf(var + LN_EPS);
        for (int k = lane; k < HIDDEN; k += 32) {
            float v = (xnbuf[r * HIDDEN + k] - mu) * inv * gamma[k] + beta[k];
            xnbuf[r * HIDDEN + k] = v;
        }
    }

    float* rd    = xnbuf;
    float* wr    = bufA;
    float* spare = bufB;

    for (int s = 0; s < 4; s++) {
        __syncthreads();
        // ---- block-diagonal matmul: wave owns blocks 4w..4w+3 ----
        // Y[16, b*32:(b+1)*32] = X[16, b*32:(b+1)*32] (16x32) @ W[s][b] (32x32)
        // One v_wmma_f32_16x16x32_f16 per 16x16 output tile (K=32 in one op).
        #pragma unroll
        for (int bi = 0; bi < 4; bi++) {
            const int b = wave * 4 + bi;
            // A operand (16x32 f16): ISA 7.12.2 16-bit A layout.
            v16h a;
            {
                const float* arow = rd + (lane & 15) * HIDDEN + b * BLOCK;
                const int kb = (lane < 16) ? 0 : 8;
                #pragma unroll
                for (int i = 0; i < 8; i++) a[i]     = (_Float16)arow[kb + i];
                #pragma unroll
                for (int i = 0; i < 8; i++) a[8 + i] = (_Float16)arow[16 + kb + i];
            }
            #pragma unroll
            for (int tn = 0; tn < 2; tn++) {
                // B operand: one contiguous 32-byte per-lane load (pre-packed).
                const int g = (s * NUM_BLOCK + b) * 2 + tn;
                const v16h bm =
                    *(const v16h*)(wpack + ((size_t)g * 32 + lane) * 16);
                v8f c = {};
                c = __builtin_amdgcn_wmma_f32_16x16x32_f16(
                        /*neg_a=*/false, a, /*neg_b=*/false, bm,
                        /*c_mod=*/(short)0, c, /*reuse_a=*/false, /*reuse_b=*/false);
                // C/D layout: VGPR r -> M = r (+8 for lanes 16-31), N = lane&15.
                const int ncol = (lane & 15) + tn * 16;
                const int mb   = (lane < 16) ? 0 : 8;
                #pragma unroll
                for (int rv = 0; rv < 8; rv++)
                    wr[(rv + mb) * HIDDEN + b * BLOCK + ncol] = c[rv];
            }
        }
        // ---- FWHT over 1024 per row, in place on wr ----
        for (int h = 1; h < HIDDEN; h <<= 1) {
            __syncthreads();
            for (int idx = tid; idx < ROWS_TILE * (HIDDEN / 2); idx += 256) {
                const int  rrow = idx >> 9;
                const int  p    = idx & 511;
                const int  i    = ((p & ~(h - 1)) << 1) | (p & (h - 1));
                float* base = wr + rrow * HIDDEN;
                const float va = base[i], vb = base[i + h];
                base[i]     = va + vb;
                base[i + h] = va - vb;
            }
        }
        __syncthreads();
        // rotate buffers; xnbuf only ever read
        if (s == 0) { rd = wr; wr = spare; }
        else        { float* t = rd; rd = wr; wr = t; }
    }

    // ---- z = 0.7*bh4 + 0.3*xn (+bias) on first 320 cols -> code & score ----
    const float* fin = rd;
    for (int pair = tid; pair < ROWS_TILE * NUM_TABLE; pair += 256) {
        const int r = pair >> 5;
        const int t = pair & 31;
        int   code  = 0;
        float score = 1.0f;
        #pragma unroll
        for (int j = 0; j < CODE_LEN; j++) {
            const int col = t * CODE_LEN + j;
            const float z = DECAY * fin[r * HIDDEN + col]
                          + (1.0f - DECAY) * xnbuf[r * HIDDEN + col]
                          + bias[col];
            code |= (z > 0.0f) ? (1 << j) : 0;
            score *= 1.0f / (1.0f + __expf(-fabsf(z)));
        }
        const long n = row0 + r;
        d_code[n * NUM_TABLE + t]  = t * TABLE_SIZE + code;   // fold table base
        d_score[n * NUM_TABLE + t] = score;
    }
}

// ---------------------------------------------------------------------------
// Kernel 2: out[n,:] = sum_t score[n,t] * tables_flat[code[n,t], :] + out_bias
// One row per workgroup; 128-bit streaming loads from the L2-resident tables,
// with global_prefetch_b8 of the next gathered row.
// ---------------------------------------------------------------------------
__global__ __launch_bounds__(256) void lookupffn_stage2(
    const float* __restrict__ tables,   // (32*1024, 1024) flat
    const int*   __restrict__ codes,    // (N, 32), table base folded in
    const float* __restrict__ scores,   // (N, 32)
    const float* __restrict__ out_bias, // (1024,)
    float*       __restrict__ out)      // (N, 1024)
{
    __shared__ int   s_code[NUM_TABLE];
    __shared__ float s_score[NUM_TABLE];
    const int  tid = threadIdx.x;      // 256 threads, 4 floats each
    const long n   = blockIdx.x;

    if (tid < NUM_TABLE) {
        s_code[tid]  = codes[n * NUM_TABLE + tid];
        s_score[tid] = scores[n * NUM_TABLE + tid];
    }
    __syncthreads();

    float4 acc = ((const float4*)out_bias)[tid];
    #pragma unroll 4
    for (int t = 0; t < NUM_TABLE; t++) {
        const float4* rowp = (const float4*)(tables + (size_t)s_code[t] * OUT_SIZE);
        if (t + 1 < NUM_TABLE)
            __builtin_prefetch(tables + (size_t)s_code[t + 1] * OUT_SIZE + tid * 4, 0, 0);
        const float4 v  = rowp[tid];
        const float  sc = s_score[t];
        acc.x += sc * v.x; acc.y += sc * v.y;
        acc.z += sc * v.z; acc.w += sc * v.w;
    }
    ((float4*)out)[n * (OUT_SIZE / 4) + tid] = acc;
}

// ---------------------------------------------------------------------------
extern "C" void kernel_launch(void* const* d_in, const int* in_sizes, int n_in,
                              void* d_out, int out_size, void* d_ws, size_t ws_size,
                              hipStream_t stream) {
    const float* x        = (const float*)d_in[0]; // hidden_states (4,2048,1024)
    const float* ln_gamma = (const float*)d_in[1];
    const float* ln_beta  = (const float*)d_in[2];
    const float* bh4_w    = (const float*)d_in[3]; // (1,4,32,32,32)
    const float* bh4_bias = (const float*)d_in[4]; // (320,)
    const float* tables   = (const float*)d_in[5]; // (32,1024,1024)
    const float* out_bias = (const float*)d_in[6]; // (1024,)
    float*       out      = (float*)d_out;

    const int N = in_sizes[0] / HIDDEN;            // 8192 rows

    // workspace layout: codes | scores | packed f16 weights
    int*      d_code  = (int*)d_ws;                                  // N*32
    float*    d_score = (float*)(d_code + (size_t)N * NUM_TABLE);    // N*32
    _Float16* d_wpack = (_Float16*)(d_score + (size_t)N * NUM_TABLE);// 128K halves

    const size_t shmem = (size_t)48 * 1024 * sizeof(float);          // 192 KB
    (void)hipFuncSetAttribute(reinterpret_cast<const void*>(lookupffn_stage1),
                              hipFuncAttributeMaxDynamicSharedMemorySize,
                              (int)shmem);

    lookupffn_pack_w<<<dim3(4 * NUM_BLOCK * 2), dim3(32), 0, stream>>>(
        bh4_w, d_wpack);

    lookupffn_stage1<<<dim3(N / ROWS_TILE), dim3(256), shmem, stream>>>(
        x, ln_gamma, ln_beta, d_wpack, bh4_bias, d_code, d_score);

    lookupffn_stage2<<<dim3(N), dim3(256), 0, stream>>>(
        tables, d_code, d_score, out_bias, out);
}
